// LocalAttention_23235773071561
// MI455X (gfx1250) — compile-verified, matrix operand order checked by reading, not run on hardware
//
#include <hip/hip_runtime.h>
#include <hip/hip_bf16.h>

// ---------------------------------------------------------------------------
// Local attention (MQA, windowed) for MI455X / gfx1250, bf16 WMMA pipeline.
//   b=1, L=4096, d_model=1024, heads=16, d_head=64, w=512, window=3w=1536.
// All matmuls run on v_wmma_f32_16x16x32_bf16 (wave32).
// ---------------------------------------------------------------------------

typedef __bf16 bf16_t;
typedef __attribute__((ext_vector_type(16))) __bf16 v16bf;
typedef __attribute__((ext_vector_type(8)))  __bf16 v8bf;
typedef __attribute__((ext_vector_type(8)))  float  v8f;
typedef __attribute__((ext_vector_type(4)))  float  v4f;

#define SEQ_L   4096
#define DMODEL  1024
#define NHEAD   16
#define DHEAD   64
#define WCH     512               // chunk width w
#define PADK    (SEQ_L + 2*WCH)   // 5120 padded key positions

static __device__ __forceinline__ bf16_t f2bf(float f) {
  unsigned u = __builtin_bit_cast(unsigned, f);
  unsigned r = (u + 0x7FFFu + ((u >> 16) & 1u)) >> 16;   // round-nearest-even
  return __builtin_bit_cast(bf16_t, (unsigned short)r);
}

static __device__ __forceinline__ v8f vzero8() {
  v8f z = {0.f,0.f,0.f,0.f,0.f,0.f,0.f,0.f};
  return z;
}

// 16x32 bf16 A/B fragment (wave32 layout): this lane reads 8 contiguous bf16
// at k = kbase+h0 and 8 more at kbase+h0+16, h0 = (lane<16 ? 0 : 8).
static __device__ __forceinline__ v16bf load_frag_bf(const bf16_t* rowptr, int kbase, int h0) {
  v8bf lo = *(const v8bf*)(rowptr + kbase + h0);        // global_load_b128
  v8bf hi = *(const v8bf*)(rowptr + kbase + h0 + 16);   // global_load_b128
  return __builtin_shufflevector(lo, hi, 0,1,2,3,4,5,6,7,8,9,10,11,12,13,14,15);
}

// Same fragment but sourced from fp32 memory (convert on the fly).
static __device__ __forceinline__ v16bf load_frag_f32(const float* rowptr, int kbase, int h0) {
  v4f a = *(const v4f*)(rowptr + kbase + h0);
  v4f b = *(const v4f*)(rowptr + kbase + h0 + 4);
  v4f c = *(const v4f*)(rowptr + kbase + h0 + 16);
  v4f d = *(const v4f*)(rowptr + kbase + h0 + 20);
  v16bf r;
#pragma unroll
  for (int i = 0; i < 4; ++i) {
    r[i]      = f2bf(a[i]);
    r[4 + i]  = f2bf(b[i]);
    r[8 + i]  = f2bf(c[i]);
    r[12 + i] = f2bf(d[i]);
  }
  return r;
}

static __device__ __forceinline__ v8f wmma_bf(v16bf a, v16bf b, v8f c) {
  return __builtin_amdgcn_wmma_f32_16x16x32_bf16(false, a, false, b, (short)0, c, false, false);
}

// ---------------------------------------------------------------------------
// Tiled WMMA GEMM:  C(MxN) = A(MxK) * Bt(NxK)^T,  Bt pre-transposed bf16.
// One wave computes a 32x64 block (2 row-strips x 4 col-tiles = 8 accum
// tiles); each B fragment is reused by both row-strips -> 8 WMMAs per k-step
// for 6 fragment loads, halving B-side L2 traffic vs a 16x64 tile.
// MODE 0: out0 = bf16 C * scale            (qp projection)
// MODE 1: cols 0..63 -> kpad (row-major, +WCH pad offset),
//         cols 64..127 -> vtpad TRANSPOSED (d x PADK)      (kv projection)
// MODE 2: out0 = fp32 C                     (final Wo projection)
// ---------------------------------------------------------------------------
template <typename AT, int MODE>
__global__ __launch_bounds__(256) void gemm_wmma(
    const AT* __restrict__ A, const bf16_t* __restrict__ Bt,
    void* __restrict__ out0, void* __restrict__ out1,
    int M, int N, int K, float scale)
{
  const int wave = blockIdx.x * 8 + (threadIdx.x >> 5);
  const int lane = threadIdx.x & 31;
  const int wavesPerRow = N >> 6;
  const int mt = wave / wavesPerRow;     // 32-row strip
  const int nb = wave % wavesPerRow;     // 64-col block
  if (mt * 32 >= M) return;

  const int col = lane & 15;
  const int h0  = (lane >> 4) ? 8 : 0;

  const AT* arow0 = A + (size_t)(mt * 32 + col) * K;
  const AT* arow1 = A + (size_t)(mt * 32 + 16 + col) * K;
  const bf16_t* brow[4];
#pragma unroll
  for (int t = 0; t < 4; ++t)
    brow[t] = Bt + (size_t)(nb * 64 + t * 16 + col) * K;

  v8f acc[2][4];
#pragma unroll
  for (int s = 0; s < 2; ++s)
#pragma unroll
    for (int t = 0; t < 4; ++t) acc[s][t] = vzero8();

  for (int k = 0; k < K; k += 32) {
    v16bf a0, a1;
    if constexpr (sizeof(AT) == 4) {
      a0 = load_frag_f32((const float*)arow0, k, h0);
      a1 = load_frag_f32((const float*)arow1, k, h0);
    } else {
      a0 = load_frag_bf((const bf16_t*)arow0, k, h0);
      a1 = load_frag_bf((const bf16_t*)arow1, k, h0);
    }
#pragma unroll
    for (int t = 0; t < 4; ++t) {
      v16bf bfrag = load_frag_bf(brow[t], k, h0);   // reused by both strips
      acc[0][t] = wmma_bf(a0, bfrag, acc[0][t]);
      acc[1][t] = wmma_bf(a1, bfrag, acc[1][t]);
    }
  }

  // D layout: lane -> (row = h0 + r within strip, col = lane&15 within n-tile)
#pragma unroll
  for (int s = 0; s < 2; ++s) {
#pragma unroll
    for (int t = 0; t < 4; ++t) {
#pragma unroll
      for (int r = 0; r < 8; ++r) {
        const int orow = mt * 32 + s * 16 + h0 + r;
        const int ocol = nb * 64 + t * 16 + col;
        const float v = acc[s][t][r] * scale;
        if constexpr (MODE == 0) {
          ((bf16_t*)out0)[(size_t)orow * N + ocol] = f2bf(v);
        } else if constexpr (MODE == 1) {
          if (ocol < DHEAD)   // K head -> kpad[WCH + pos][d]
            ((bf16_t*)out0)[(size_t)(WCH + orow) * DHEAD + ocol] = f2bf(v);
          else                // V head -> vtpad[d][WCH + pos]  (transposed)
            ((bf16_t*)out1)[(size_t)(ocol - DHEAD) * PADK + (WCH + orow)] = f2bf(v);
        } else {
          ((float*)out0)[(size_t)orow * N + ocol] = v;
        }
      }
    }
  }
}

// ---------------------------------------------------------------------------
// Windowed MQA attention, one wave per (chunk, head, 16-query strip).
// Computes ctx^T = V^T . softmax(K . Q^T) with online (flash) softmax.
// The D-layout of S' = K.Q^T IS the B-fragment layout of P' for the V^T WMMA,
// so no LDS / no shuffles are needed for the "transpose".
// ---------------------------------------------------------------------------
__global__ __launch_bounds__(256) void attn_wmma(
    const bf16_t* __restrict__ qp,     // (L, 1024) bf16, already scaled by 1/8
    const bf16_t* __restrict__ kpad,   // (PADK, 64) bf16, zero-padded chunks
    const bf16_t* __restrict__ vtp,    // (64, PADK) bf16 transposed, padded
    bf16_t* __restrict__ ctx)          // (L, 1024) bf16
{
  const int wave  = blockIdx.x * 8 + (threadIdx.x >> 5);
  const int lane  = threadIdx.x & 31;
  const int strip = wave & 31;          // 32 strips of 16 queries per chunk
  const int head  = (wave >> 5) & 15;
  const int chunk = wave >> 9;          // 8 chunks

  const int col = lane & 15;            // query column in all B/D fragments
  const int h0  = (lane >> 4) ? 8 : 0;

  const int qbase = chunk * WCH + strip * 16;
  const bf16_t* qrow = qp + (size_t)(qbase + col) * DMODEL + head * DHEAD;
  const v16bf bq0 = load_frag_bf(qrow, 0, h0);    // Q^T B-fragments, d 0..31
  const v16bf bq1 = load_frag_bf(qrow, 32, h0);   //                  d 32..63

  v8f acc[4] = {vzero8(), vzero8(), vzero8(), vzero8()};  // ctx^T, 4 d-tiles
  float m = -1e30f, l = 0.f;

  const int kwin = chunk * WCH;  // window start in padded key coordinates

  for (int kt = 0; kt < 3 * WCH; kt += 32) {
    // K tile A-fragments (keys x d), rows contiguous in kpad
    const bf16_t* k0row = kpad + (size_t)(kwin + kt + col) * DHEAD;
    const bf16_t* k1row = kpad + (size_t)(kwin + kt + 16 + col) * DHEAD;
    v16bf ak00 = load_frag_bf(k0row, 0, h0);
    v16bf ak01 = load_frag_bf(k0row, 32, h0);
    v16bf ak10 = load_frag_bf(k1row, 0, h0);
    v16bf ak11 = load_frag_bf(k1row, 32, h0);

    // prefetch next 32-key tile of K and V (global_prefetch_b8)
    if (kt + 32 < 3 * WCH) {
      __builtin_prefetch(kpad + (size_t)(kwin + kt + 32 + col) * DHEAD, 0, 1);
      __builtin_prefetch(vtp + (size_t)col * PADK + kwin + kt + 32, 0, 1);
    }

    v8f s0 = wmma_bf(ak01, bq1, wmma_bf(ak00, bq0, vzero8()));  // keys 0..15
    v8f s1 = wmma_bf(ak11, bq1, wmma_bf(ak10, bq0, vzero8()));  // keys 16..31

    // per-query (column) max: 16 regs in-lane + exchange with lane^16
    float mt = s0[0];
#pragma unroll
    for (int r = 1; r < 8; ++r) mt = fmaxf(mt, s0[r]);
#pragma unroll
    for (int r = 0; r < 8; ++r) mt = fmaxf(mt, s1[r]);
    mt = fmaxf(mt, __shfl_xor(mt, 16, 32));

    const float mn = fmaxf(m, mt);
    const float rs = __expf(m - mn);
    m = mn;

    float ps = 0.f;
    v16bf bp;                       // P' B-fragment built in-place from D regs
#pragma unroll
    for (int r = 0; r < 8; ++r) {
      const float p0 = __expf(s0[r] - mn);
      const float p1 = __expf(s1[r] - mn);
      ps += p0 + p1;
      bp[r]     = f2bf(p0);         // keys h0+r       -> elements 0..7
      bp[8 + r] = f2bf(p1);         // keys 16+h0+r    -> elements 8..15
    }
    ps += __shfl_xor(ps, 16, 32);
    l = l * rs + ps;

#pragma unroll
    for (int t = 0; t < 4; ++t)
#pragma unroll
      for (int r = 0; r < 8; ++r) acc[t][r] *= rs;

    // ctx^T += V^T(tile) . P'   (A-fragments contiguous in transposed V)
#pragma unroll
    for (int t = 0; t < 4; ++t) {
      const bf16_t* vrow = vtp + (size_t)(t * 16 + col) * PADK + kwin + kt;
      v16bf av = load_frag_bf(vrow, 0, h0);
      acc[t] = wmma_bf(av, bp, acc[t]);
    }
  }

  const float inv = 1.f / l;
  const size_t opos = (size_t)(qbase + col) * DMODEL + head * DHEAD;
#pragma unroll
  for (int t = 0; t < 4; ++t)
#pragma unroll
    for (int r = 0; r < 8; ++r)
      ctx[opos + t * 16 + h0 + r] = f2bf(acc[t][r] * inv);
}

// ---------------------------------------------------------------------------
__global__ __launch_bounds__(256) void transpose_to_bf16(
    const float* __restrict__ in, bf16_t* __restrict__ out, int K, int N)
{
  const int idx = blockIdx.x * 256 + threadIdx.x;
  if (idx >= K * N) return;
  const int k = idx / N, n = idx % N;
  out[(size_t)n * K + k] = f2bf(in[idx]);
}

__global__ __launch_bounds__(256) void zero_u4(uint4* __restrict__ p, int n) {
  const int i = blockIdx.x * 256 + threadIdx.x;
  if (i < n) p[i] = make_uint4(0u, 0u, 0u, 0u);
}

// ---------------------------------------------------------------------------
extern "C" void kernel_launch(void* const* d_in, const int* in_sizes, int n_in,
                              void* d_out, int out_size, void* d_ws, size_t ws_size,
                              hipStream_t stream)
{
  const float* q   = (const float*)d_in[0];
  const float* kv  = (const float*)d_in[1];
  const float* Wq  = (const float*)d_in[2];
  const float* Wkv = (const float*)d_in[3];
  const float* Wo  = (const float*)d_in[4];
  // d_in[5] = w (== 512, baked into WCH)

  char* ws = (char*)d_ws;
  size_t off = 0;
  bf16_t* qp   = (bf16_t*)(ws + off); off += (size_t)SEQ_L * DMODEL * 2;   // 8 MB
  bf16_t* kpad = (bf16_t*)(ws + off); off += (size_t)PADK * DHEAD * 2;     // 640 KB
  bf16_t* vtp  = (bf16_t*)(ws + off); off += (size_t)DHEAD * PADK * 2;     // 640 KB
  bf16_t* ctxb = (bf16_t*)(ws + off); off += (size_t)SEQ_L * DMODEL * 2;   // 8 MB
  bf16_t* Wqt  = (bf16_t*)(ws + off); off += (size_t)DMODEL * DMODEL * 2;  // 2 MB
  bf16_t* Wkvt = (bf16_t*)(ws + off); off += (size_t)DMODEL * 128 * 2;     // 256 KB
  bf16_t* Wot  = (bf16_t*)(ws + off); off += (size_t)DMODEL * DMODEL * 2;  // 2 MB

  // 1) weights -> transposed bf16 (so every WMMA B fragment is contiguous)
  transpose_to_bf16<<<(DMODEL * DMODEL + 255) / 256, 256, 0, stream>>>(Wq, Wqt, DMODEL, DMODEL);
  transpose_to_bf16<<<(DMODEL * 128 + 255) / 256, 256, 0, stream>>>(Wkv, Wkvt, DMODEL, 128);
  transpose_to_bf16<<<(DMODEL * DMODEL + 255) / 256, 256, 0, stream>>>(Wo, Wot, DMODEL, DMODEL);

  // 2) zero K/V pads (kpad & vtp are contiguous) -> reference's zero-score slots
  const int nz = (2 * PADK * DHEAD * 2) / 16;
  zero_u4<<<(nz + 255) / 256, 256, 0, stream>>>((uint4*)kpad, nz);

  // 3) qp = (q @ Wq) / 8, bf16      : 4096x1024x1024, 32x64 tiles -> 2048 waves
  gemm_wmma<float, 0><<<256, 256, 0, stream>>>(q, Wqt, qp, nullptr,
                                               SEQ_L, DMODEL, DMODEL, 0.125f);
  // 4) kv @ Wkv -> kpad / vtp(T)    : 4096x128x1024 -> 256 waves
  gemm_wmma<float, 1><<<32, 256, 0, stream>>>(kv, Wkvt, kpad, vtp,
                                              SEQ_L, 128, DMODEL, 1.0f);
  // 5) windowed attention: 8 chunks x 16 heads x 32 strips = 4096 waves
  attn_wmma<<<512, 256, 0, stream>>>(qp, kpad, vtp, ctxb);

  // 6) out = ctx @ Wo, fp32         : 4096x1024x1024 -> 2048 waves
  gemm_wmma<bf16_t, 2><<<256, 256, 0, stream>>>(ctxb, Wot, d_out, nullptr,
                                                SEQ_L, DMODEL, DMODEL, 1.0f);
}
